// UrbanGNN_47364899340608
// MI455X (gfx1250) — matrix operand order repaired and checked by reading, not compiled, not run on hardware
//
#include <hip/hip_runtime.h>
#include <math.h>

#define N_NODES 20000
#define N_EDGES 320000
#define N_GRAPH 8
#define HC_DIM  256
#define HID_DIM 64
#define NEG_SLOPE 0.2f

typedef __attribute__((ext_vector_type(16))) _Float16 v16h;
typedef __attribute__((ext_vector_type(8)))  float    v8f;

static __device__ __forceinline__ void atomicMaxF(float* addr, float val) {
  unsigned int* ua = (unsigned int*)addr;
  unsigned int old = *ua;
  while (val > __uint_as_float(old)) {
    unsigned int assumed = old;
    old = atomicCAS(ua, assumed, __float_as_uint(val));
    if (old == assumed) break;
  }
}

__global__ void fill_kernel(float* p, float v, long n) {
  long i = (long)blockIdx.x * blockDim.x + threadIdx.x;
  if (i < n) p[i] = v;
}

// ---- pad x: (N x 18) -> (N x 32), zero-filled tail ----
__global__ void pad_x(const float* __restrict__ x, float* __restrict__ xp, int Nn) {
  int i = blockIdx.x * blockDim.x + threadIdx.x;
  if (i >= Nn * 32) return;
  int n = i >> 5, k = i & 31;
  xp[i] = (k < 18) ? x[(size_t)n * 18 + k] : 0.0f;
}

// ---- pack W (Kreal x Nc, f32) -> B-fragment-ordered f16, Kpad%32==0 ----
// layout: (((ntile * nkc + kc) * 32 + lane) * 16 + i) holds W[kc*32 + (lane>>4)*16 + i][ntile*16 + (lane&15)]
__global__ void pack_w(const float* __restrict__ W, _Float16* __restrict__ Wp,
                       int Kpad, int Kreal, int Nc) {
  int idx = blockIdx.x * blockDim.x + threadIdx.x;
  if (idx >= Kpad * Nc) return;
  int i    = idx & 15;
  int lane = (idx >> 4) & 31;
  int t    = idx >> 9;
  int nkc  = Kpad >> 5;
  int kc   = t % nkc;
  int nt   = t / nkc;
  int k = kc * 32 + (lane >> 4) * 16 + i;
  int n = nt * 16 + (lane & 15);
  Wp[idx] = (_Float16)((k < Kreal) ? W[(size_t)k * Nc + n] : 0.0f);
}

// ---- self-loop attr: cnt[dst]++, lat[dst] += ea[e]; then lat /= max(cnt,1) ----
__global__ void loop_scatter(const float* __restrict__ ea, const int* __restrict__ dst,
                             float* cnt, float* lat, int E) {
  int e = blockIdx.x * blockDim.x + threadIdx.x;
  if (e >= E) return;
  int d = dst[e];
  float4 a = *(const float4*)(ea + (size_t)e * 4);
  atomicAdd(&cnt[d], 1.0f);
  atomicAdd(&lat[d * 4 + 0], a.x);
  atomicAdd(&lat[d * 4 + 1], a.y);
  atomicAdd(&lat[d * 4 + 2], a.z);
  atomicAdd(&lat[d * 4 + 3], a.w);
}
__global__ void loop_final(float* lat, const float* cnt, int Nn) {
  int i = blockIdx.x * blockDim.x + threadIdx.x;
  if (i >= Nn * 4) return;
  lat[i] /= fmaxf(cnt[i >> 2], 1.0f);
}

// ---- WMMA GEMM, K%32==0, W pre-packed f16 fragments ----
// grid: (Nc/16, ceil(M/64)), block 128 (4 waves, one 16x16 tile each along M)
__global__ void wmma_gemm(const float* __restrict__ A, const _Float16* __restrict__ Wp,
                          const float* __restrict__ bias, float* __restrict__ C,
                          int M, int K, int Nc, int act, int acc) {
  int lane  = threadIdx.x & 31;
  int wv    = threadIdx.x >> 5;
  int mtile = blockIdx.y * 4 + wv;
  if (mtile * 16 >= M) return;                 // wave-uniform: EXEC stays all-ones
  int mbase = mtile * 16;
  int nbase = blockIdx.x * 16;
  int mrow  = mbase + (lane & 15);
  int ncol  = nbase + (lane & 15);
  int khalf = lane >> 4;
  int nkc   = K >> 5;
  const float*    arow = A + (size_t)mrow * K + khalf * 16;
  const _Float16* wb   = Wp + ((size_t)blockIdx.x * nkc * 32 + lane) * 16;
  v8f c = {0.f, 0.f, 0.f, 0.f, 0.f, 0.f, 0.f, 0.f};
  for (int kc = 0; kc < nkc; ++kc) {
    const float* ap = arow + kc * 32;
    float4 q0 = *(const float4*)(ap);
    float4 q1 = *(const float4*)(ap + 4);
    float4 q2 = *(const float4*)(ap + 8);
    float4 q3 = *(const float4*)(ap + 12);
    v16h b = *(const v16h*)(wb + (size_t)kc * 512);
    v16h a;
    a[0] = (_Float16)q0.x; a[1] = (_Float16)q0.y; a[2]  = (_Float16)q0.z; a[3]  = (_Float16)q0.w;
    a[4] = (_Float16)q1.x; a[5] = (_Float16)q1.y; a[6]  = (_Float16)q1.z; a[7]  = (_Float16)q1.w;
    a[8] = (_Float16)q2.x; a[9] = (_Float16)q2.y; a[10] = (_Float16)q2.z; a[11] = (_Float16)q2.w;
    a[12]= (_Float16)q3.x; a[13]= (_Float16)q3.y; a[14] = (_Float16)q3.z; a[15] = (_Float16)q3.w;
    c = __builtin_amdgcn_wmma_f32_16x16x32_f16(false, a, false, b, (short)0, c, false, false);
  }
  int mo = khalf * 8;
#pragma unroll
  for (int r = 0; r < 8; ++r) {
    size_t idx = (size_t)(mbase + mo + r) * Nc + ncol;
    float v = c[r];
    if (acc) v += C[idx];
    if (bias) v += bias[ncol];
    if (act == 1) v = fmaxf(v, 0.0f);
    C[idx] = v;
  }
}

// ---- GATv2 pass 1: per-edge logits + segment max. wave per edge, lane owns 8 chans ----
__global__ void gat_logit(const float* __restrict__ xl, const float* __restrict__ xr,
                          const float* __restrict__ ea, const float* __restrict__ lat,
                          const int* __restrict__ srcv, const int* __restrict__ dstv,
                          const float* __restrict__ we, const float* __restrict__ att,
                          float* logit, float* mx, int E, int Nn) {
  __shared__ float s_we[4 * HC_DIM];
  __shared__ float s_att[HC_DIM];
  for (int i = threadIdx.x; i < 4 * HC_DIM; i += blockDim.x) s_we[i] = we[i];
  for (int i = threadIdx.x; i < HC_DIM; i += blockDim.x) s_att[i] = att[i];
  __syncthreads();
  int wv = threadIdx.x >> 5, lane = threadIdx.x & 31;
  int e = blockIdx.x * 8 + wv;
  if (e >= E + Nn) return;
  int s, d; float4 aea;
  if (e < E) { s = srcv[e]; d = dstv[e]; aea = *(const float4*)(ea + (size_t)e * 4); }
  else       { s = d = e - E;            aea = *(const float4*)(lat + (size_t)(e - E) * 4); }
  const float* pl = xl + (size_t)s * HC_DIM + lane * 8;
  const float* pr = xr + (size_t)d * HC_DIM + lane * 8;
  float4 l0 = *(const float4*)(pl);
  float4 l1 = *(const float4*)(pl + 4);
  float4 r0 = *(const float4*)(pr);
  float4 r1 = *(const float4*)(pr + 4);
  float lv[8] = {l0.x, l0.y, l0.z, l0.w, l1.x, l1.y, l1.z, l1.w};
  float rv[8] = {r0.x, r0.y, r0.z, r0.w, r1.x, r1.y, r1.z, r1.w};
  float part = 0.0f;
#pragma unroll
  for (int i = 0; i < 8; ++i) {
    int dd = lane * 8 + i;
    float ee = aea.x * s_we[dd] + aea.y * s_we[HC_DIM + dd] +
               aea.z * s_we[2 * HC_DIM + dd] + aea.w * s_we[3 * HC_DIM + dd];
    float m = lv[i] + rv[i] + ee;
    m = (m > 0.0f) ? m : NEG_SLOPE * m;          // leaky_relu
    part += m * s_att[dd];
  }
  part += __shfl_xor(part, 1, 32);
  part += __shfl_xor(part, 2, 32);
  part += __shfl_xor(part, 4, 32);
  if ((lane & 7) == 0) {
    int h = lane >> 3;
    logit[(size_t)e * 4 + h] = part;
    atomicMaxF(&mx[(size_t)d * 4 + h], part);
  }
}

// ---- GATv2 pass 2: ex = exp(logit - mx[dst]); den[dst] += ex ----
__global__ void gat_expsum(const int* __restrict__ dstv, float* logit,
                           const float* __restrict__ mx, float* den, int E, int Nn) {
  int e = blockIdx.x * blockDim.x + threadIdx.x;
  if (e >= E + Nn) return;
  int d = (e < E) ? dstv[e] : (e - E);
#pragma unroll
  for (int h = 0; h < 4; ++h) {
    float ex = expf(logit[(size_t)e * 4 + h] - mx[(size_t)d * 4 + h]);
    logit[(size_t)e * 4 + h] = ex;
    atomicAdd(&den[(size_t)d * 4 + h], ex);
  }
}

// ---- GATv2 pass 3: out[dst] += (ex/den[dst]) * xl[src]; wave per edge ----
__global__ void gat_scatter(const float* __restrict__ xl, const float* __restrict__ logit,
                            const float* __restrict__ den, const int* __restrict__ srcv,
                            const int* __restrict__ dstv, float* out, int E, int Nn) {
  int wv = threadIdx.x >> 5, lane = threadIdx.x & 31;
  int e = blockIdx.x * 8 + wv;
  if (e >= E + Nn) return;
  int s, d;
  if (e < E) { s = srcv[e]; d = dstv[e]; } else { s = d = e - E; }
  int h = lane >> 3;
  float alpha = logit[(size_t)e * 4 + h] / den[(size_t)d * 4 + h];
  const float* pl = xl + (size_t)s * HC_DIM + lane * 8;
  float* po = out + (size_t)d * HC_DIM + lane * 8;
  float4 l0 = *(const float4*)(pl);
  float4 l1 = *(const float4*)(pl + 4);
  float lv[8] = {l0.x, l0.y, l0.z, l0.w, l1.x, l1.y, l1.z, l1.w};
#pragma unroll
  for (int i = 0; i < 8; ++i) atomicAdd(&po[i], alpha * lv[i]);
}

// ---- finalize GAT1: h1 = elu(bn1(out + g1_b)) in place (N*256) ----
__global__ void gat1_final(float* out, const float* __restrict__ bias,
                           const float* __restrict__ g, const float* __restrict__ b, long total) {
  long i = (long)blockIdx.x * blockDim.x + threadIdx.x;
  if (i >= total) return;
  int dd = (int)(i & (HC_DIM - 1));
  float rs = rsqrtf(1.0f + 1e-5f);
  float v = (out[i] + bias[dd]) * (g[dd] * rs) + b[dd];
  out[i] = (v > 0.0f) ? v : (expf(v) - 1.0f);
}

// ---- finalize GAT2: h2 = elu(bn2(mean_heads(out2) + g2_b)) -> (N*64) ----
__global__ void gat2_final(const float* __restrict__ out2, const float* __restrict__ bias,
                           const float* __restrict__ g, const float* __restrict__ b,
                           float* h2, int Nn) {
  int i = blockIdx.x * blockDim.x + threadIdx.x;
  if (i >= Nn * HID_DIM) return;
  int n = i >> 6, c = i & 63;
  const float* p = out2 + (size_t)n * HC_DIM + c;
  float rs = rsqrtf(1.0f + 1e-5f);
  float v = (p[0] + p[64] + p[128] + p[192]) * 0.25f + bias[c];
  v = v * (g[c] * rs) + b[c];
  h2[i] = (v > 0.0f) ? v : (expf(v) - 1.0f);
}

// ---- SAGE mean aggregation ----
__global__ void sage_scatter(const float* __restrict__ h2, const int* __restrict__ srcv,
                             const int* __restrict__ dstv, float* agg, int E) {
  int wv = threadIdx.x >> 5, lane = threadIdx.x & 31;
  int e = blockIdx.x * 8 + wv;
  if (e >= E) return;
  int s = srcv[e], d = dstv[e];
  const float* ph = h2 + (size_t)s * HID_DIM;
  float* pa = agg + (size_t)d * HID_DIM;
  atomicAdd(&pa[lane], ph[lane]);
  atomicAdd(&pa[lane + 32], ph[lane + 32]);
}
__global__ void sage_div(float* agg, const float* __restrict__ cnt, int Nn) {
  int i = blockIdx.x * blockDim.x + threadIdx.x;
  if (i >= Nn * HID_DIM) return;
  agg[i] /= fmaxf(cnt[i >> 6], 1.0f);
}

// ---- BN3 + relu in place (N*64) ----
__global__ void bn3_relu(float* h, const float* __restrict__ g, const float* __restrict__ b, int Nn) {
  int i = blockIdx.x * blockDim.x + threadIdx.x;
  if (i >= Nn * HID_DIM) return;
  int c = i & 63;
  float rs = rsqrtf(1.0f + 1e-5f);
  h[i] = fmaxf(h[i] * (g[c] * rs) + b[c], 0.0f);
}

// ---- global mean+max pooling; wave per node ----
__global__ void pool_kernel(const float* __restrict__ h, const int* __restrict__ batch,
                            float* mean_p, float* max_p, float* cntg, int Nn) {
  int wv = threadIdx.x >> 5, lane = threadIdx.x & 31;
  int n = blockIdx.x * 8 + wv;
  if (n >= Nn) return;
  int g = batch[n];
  float v0 = h[(size_t)n * HID_DIM + lane];
  float v1 = h[(size_t)n * HID_DIM + 32 + lane];
  atomicAdd(&mean_p[g * HID_DIM + lane], v0);
  atomicAdd(&mean_p[g * HID_DIM + 32 + lane], v1);
  atomicMaxF(&max_p[g * HID_DIM + lane], v0);
  atomicMaxF(&max_p[g * HID_DIM + 32 + lane], v1);
  if (lane == 0) atomicAdd(&cntg[g], 1.0f);
}

// ---- graph head MLP (G=8, single block) ----
__global__ void graph_head(const float* __restrict__ mean_p, const float* __restrict__ max_p,
                           const float* __restrict__ cntg,
                           const float* __restrict__ w1, const float* __restrict__ b1,
                           const float* __restrict__ w2, const float* __restrict__ b2,
                           const float* __restrict__ w3, const float* __restrict__ b3,
                           float* out) {
  __shared__ float s_hp[N_GRAPH * 128];
  __shared__ float s_z1[N_GRAPH * 64];
  __shared__ float s_z2[N_GRAPH * 32];
  int t = threadIdx.x;
  for (int i = t; i < N_GRAPH * 128; i += blockDim.x) {
    int g = i >> 7, k = i & 127;
    float c = fmaxf(cntg[g], 1.0f);
    s_hp[i] = (k < 64) ? mean_p[g * 64 + k] / c : max_p[g * 64 + (k - 64)];
  }
  __syncthreads();
  for (int i = t; i < N_GRAPH * 64; i += blockDim.x) {
    int g = i >> 6, j = i & 63;
    float a = b1[j];
    for (int k = 0; k < 128; ++k) a += s_hp[g * 128 + k] * w1[k * 64 + j];
    s_z1[i] = fmaxf(a, 0.0f);
  }
  __syncthreads();
  for (int i = t; i < N_GRAPH * 32; i += blockDim.x) {
    int g = i >> 5, j = i & 31;
    float a = b2[j];
    for (int k = 0; k < 64; ++k) a += s_z1[g * 64 + k] * w2[k * 32 + j];
    s_z2[i] = fmaxf(a, 0.0f);
  }
  __syncthreads();
  if (t < N_GRAPH) {
    float a = b3[0];
    for (int k = 0; k < 32; ++k) a += s_z2[t * 32 + k] * w3[k];
    out[t] = 1.0f / (1.0f + expf(-a));
  }
}

// ---- edge head MLP fused (thread per edge; 132x32 weights in LDS) ----
__global__ void edge_head(const float* __restrict__ h, const float* __restrict__ ea,
                          const int* __restrict__ srcv, const int* __restrict__ dstv,
                          const float* __restrict__ w1, const float* __restrict__ b1,
                          const float* __restrict__ w2, const float* __restrict__ b2,
                          float* out, int E) {
  __shared__ float s_w1[132 * 32];
  __shared__ float s_b1[32];
  __shared__ float s_w2[32];
  __shared__ float s_b2;
  for (int i = threadIdx.x; i < 132 * 32; i += blockDim.x) s_w1[i] = w1[i];
  if (threadIdx.x < 32) { s_b1[threadIdx.x] = b1[threadIdx.x]; s_w2[threadIdx.x] = w2[threadIdx.x]; }
  if (threadIdx.x == 0) s_b2 = b2[0];
  __syncthreads();
  int e = blockIdx.x * blockDim.x + threadIdx.x;
  if (e >= E) return;
  int s = srcv[e], d = dstv[e];
  float z[32];
#pragma unroll
  for (int j = 0; j < 32; ++j) z[j] = s_b1[j];
  const float* ps = h + (size_t)s * HID_DIM;
#pragma unroll 4
  for (int k4 = 0; k4 < 16; ++k4) {
    float4 v = *(const float4*)(ps + k4 * 4);
    const float* w = &s_w1[k4 * 4 * 32];
#pragma unroll
    for (int j = 0; j < 32; ++j)
      z[j] += v.x * w[j] + v.y * w[32 + j] + v.z * w[64 + j] + v.w * w[96 + j];
  }
  const float* pd = h + (size_t)d * HID_DIM;
#pragma unroll 4
  for (int k4 = 0; k4 < 16; ++k4) {
    float4 v = *(const float4*)(pd + k4 * 4);
    const float* w = &s_w1[(64 + k4 * 4) * 32];
#pragma unroll
    for (int j = 0; j < 32; ++j)
      z[j] += v.x * w[j] + v.y * w[32 + j] + v.z * w[64 + j] + v.w * w[96 + j];
  }
  float4 v = *(const float4*)(ea + (size_t)e * 4);
  {
    const float* w = &s_w1[128 * 32];
#pragma unroll
    for (int j = 0; j < 32; ++j)
      z[j] += v.x * w[j] + v.y * w[32 + j] + v.z * w[64 + j] + v.w * w[96 + j];
  }
  float a = s_b2;
#pragma unroll
  for (int j = 0; j < 32; ++j) a += fmaxf(z[j], 0.0f) * s_w2[j];
  out[e] = 1.0f / (1.0f + expf(-a));
}

extern "C" void kernel_launch(void* const* d_in, const int* in_sizes, int n_in,
                              void* d_out, int out_size, void* d_ws, size_t ws_size,
                              hipStream_t stream) {
  const float* x        = (const float*)d_in[0];
  const float* edge_attr= (const float*)d_in[1];
  const float* enc_w1 = (const float*)d_in[2];  const float* enc_b1 = (const float*)d_in[3];
  const float* enc_w2 = (const float*)d_in[4];  const float* enc_b2 = (const float*)d_in[5];
  const float* g1_wl  = (const float*)d_in[6];  const float* g1_wr  = (const float*)d_in[7];
  const float* g1_we  = (const float*)d_in[8];  const float* g1_att = (const float*)d_in[9];
  const float* g1_b   = (const float*)d_in[10];
  const float* bn1_g  = (const float*)d_in[11]; const float* bn1_b  = (const float*)d_in[12];
  const float* g2_wl  = (const float*)d_in[13]; const float* g2_wr  = (const float*)d_in[14];
  const float* g2_we  = (const float*)d_in[15]; const float* g2_att = (const float*)d_in[16];
  const float* g2_b   = (const float*)d_in[17];
  const float* bn2_g  = (const float*)d_in[18]; const float* bn2_b  = (const float*)d_in[19];
  const float* sage_wl= (const float*)d_in[20]; const float* sage_bl= (const float*)d_in[21];
  const float* sage_wr= (const float*)d_in[22];
  const float* bn3_g  = (const float*)d_in[23]; const float* bn3_b  = (const float*)d_in[24];
  const float* gh_w1  = (const float*)d_in[25]; const float* gh_b1  = (const float*)d_in[26];
  const float* gh_w2  = (const float*)d_in[27]; const float* gh_b2  = (const float*)d_in[28];
  const float* gh_w3  = (const float*)d_in[29]; const float* gh_b3  = (const float*)d_in[30];
  const float* eh_w1  = (const float*)d_in[31]; const float* eh_b1  = (const float*)d_in[32];
  const float* eh_w2  = (const float*)d_in[33]; const float* eh_b2  = (const float*)d_in[34];
  const int* edge_index = (const int*)d_in[35];
  const int* batch      = (const int*)d_in[36];
  const int* srcv = edge_index;
  const int* dstv = edge_index + N_EDGES;
  float* out = (float*)d_out;

  const int Nn = N_NODES, E = N_EDGES, EP = N_EDGES + N_NODES;

  // workspace layout (floats); all offsets multiples of 16 floats (64B)
  float* ws = (float*)d_ws;
  size_t o = 0;
  float* wsH0   = ws + o; o += (size_t)Nn * 64;    // h0 / later h2
  float* wsH    = ws + o; o += (size_t)Nn * 64;    // encoder h / later agg
  float* wsXL   = ws + o; o += (size_t)Nn * 256;
  float* wsXR   = ws + o; o += (size_t)Nn * 256;
  float* wsOUT  = ws + o; o += (size_t)Nn * 256;   // gat accum / h1
  float* wsLG   = ws + o; o += (size_t)EP * 4;     // logits / unnorm alpha
  float* wsCNT  = ws + o; o += (size_t)Nn;
  float* wsLAT  = ws + o; o += (size_t)Nn * 4;
  float* wsMX   = ws + o; o += (size_t)Nn * 4;
  float* wsDEN  = ws + o; o += (size_t)Nn * 4;
  float* wsH3   = ws + o; o += (size_t)Nn * 64;
  float* wsXP   = ws + o; o += (size_t)Nn * 32;    // padded x (N x 32)
  float* wsMEAN = ws + o; o += (size_t)N_GRAPH * 64;
  float* wsMAX  = ws + o; o += (size_t)N_GRAPH * 64;
  float* wsCNTG = ws + o; o += 16;
  // packed f16 weights (sizes in halfs, kept 32B-aligned via float-16 rounding)
  _Float16* pEW1 = (_Float16*)(ws + o); o += (32   * 64 ) / 2;   // enc_w1 padded K=32
  _Float16* pEW2 = (_Float16*)(ws + o); o += (64   * 64 ) / 2;
  _Float16* pG1L = (_Float16*)(ws + o); o += (64   * 256) / 2;
  _Float16* pG1R = (_Float16*)(ws + o); o += (64   * 256) / 2;
  _Float16* pG2L = (_Float16*)(ws + o); o += (256  * 256) / 2;
  _Float16* pG2R = (_Float16*)(ws + o); o += (256  * 256) / 2;
  _Float16* pSWL = (_Float16*)(ws + o); o += (64   * 64 ) / 2;
  _Float16* pSWR = (_Float16*)(ws + o); o += (64   * 64 ) / 2;

  auto cdiv = [](long a, long b) { return (int)((a + b - 1) / b); };
  const float NEGINF = -3.4e38f;

  // ---- init + packing ----
  fill_kernel<<<cdiv(Nn, 256), 256, 0, stream>>>(wsCNT, 0.0f, Nn);
  fill_kernel<<<cdiv(Nn * 4, 256), 256, 0, stream>>>(wsLAT, 0.0f, (long)Nn * 4);
  fill_kernel<<<cdiv(Nn * 4, 256), 256, 0, stream>>>(wsMX, NEGINF, (long)Nn * 4);
  fill_kernel<<<cdiv(Nn * 4, 256), 256, 0, stream>>>(wsDEN, 0.0f, (long)Nn * 4);
  fill_kernel<<<cdiv((long)Nn * 256, 256), 256, 0, stream>>>(wsOUT, 0.0f, (long)Nn * 256);
  fill_kernel<<<cdiv(N_GRAPH * 64, 256), 256, 0, stream>>>(wsMEAN, 0.0f, N_GRAPH * 64);
  fill_kernel<<<cdiv(N_GRAPH * 64, 256), 256, 0, stream>>>(wsMAX, NEGINF, N_GRAPH * 64);
  fill_kernel<<<1, 256, 0, stream>>>(wsCNTG, 0.0f, N_GRAPH);
  pad_x<<<cdiv(Nn * 32, 256), 256, 0, stream>>>(x, wsXP, Nn);
  pack_w<<<cdiv(32 * 64, 256),   256, 0, stream>>>(enc_w1,  pEW1, 32,  18,  64);
  pack_w<<<cdiv(64 * 64, 256),   256, 0, stream>>>(enc_w2,  pEW2, 64,  64,  64);
  pack_w<<<cdiv(64 * 256, 256),  256, 0, stream>>>(g1_wl,   pG1L, 64,  64,  256);
  pack_w<<<cdiv(64 * 256, 256),  256, 0, stream>>>(g1_wr,   pG1R, 64,  64,  256);
  pack_w<<<cdiv(256 * 256, 256), 256, 0, stream>>>(g2_wl,   pG2L, 256, 256, 256);
  pack_w<<<cdiv(256 * 256, 256), 256, 0, stream>>>(g2_wr,   pG2R, 256, 256, 256);
  pack_w<<<cdiv(64 * 64, 256),   256, 0, stream>>>(sage_wl, pSWL, 64,  64,  64);
  pack_w<<<cdiv(64 * 64, 256),   256, 0, stream>>>(sage_wr, pSWR, 64,  64,  64);

  // ---- self-loop mean edge_attr ----
  loop_scatter<<<cdiv(E, 256), 256, 0, stream>>>(edge_attr, dstv, wsCNT, wsLAT, E);
  loop_final<<<cdiv(Nn * 4, 256), 256, 0, stream>>>(wsLAT, wsCNT, Nn);

  // ---- encoder: h0 = relu(x@w1+b1); h = h0@w2+b2 ----
  wmma_gemm<<<dim3(4, cdiv(Nn, 64)), 128, 0, stream>>>(wsXP, pEW1, enc_b1, wsH0, Nn, 32, 64, 1, 0);
  wmma_gemm<<<dim3(4, cdiv(Nn, 64)), 128, 0, stream>>>(wsH0, pEW2, enc_b2, wsH, Nn, 64, 64, 0, 0);

  // ---- GAT layer 1 ----
  wmma_gemm<<<dim3(16, cdiv(Nn, 64)), 128, 0, stream>>>(wsH, pG1L, nullptr, wsXL, Nn, 64, 256, 0, 0);
  wmma_gemm<<<dim3(16, cdiv(Nn, 64)), 128, 0, stream>>>(wsH, pG1R, nullptr, wsXR, Nn, 64, 256, 0, 0);
  gat_logit<<<cdiv(EP, 8), 256, 0, stream>>>(wsXL, wsXR, edge_attr, wsLAT, srcv, dstv, g1_we, g1_att, wsLG, wsMX, E, Nn);
  gat_expsum<<<cdiv(EP, 256), 256, 0, stream>>>(dstv, wsLG, wsMX, wsDEN, E, Nn);
  gat_scatter<<<cdiv(EP, 8), 256, 0, stream>>>(wsXL, wsLG, wsDEN, srcv, dstv, wsOUT, E, Nn);
  gat1_final<<<cdiv((long)Nn * 256, 256), 256, 0, stream>>>(wsOUT, g1_b, bn1_g, bn1_b, (long)Nn * 256);

  // ---- GAT layer 2 (input h1 = wsOUT) ----
  wmma_gemm<<<dim3(16, cdiv(Nn, 64)), 128, 0, stream>>>(wsOUT, pG2L, nullptr, wsXL, Nn, 256, 256, 0, 0);
  wmma_gemm<<<dim3(16, cdiv(Nn, 64)), 128, 0, stream>>>(wsOUT, pG2R, nullptr, wsXR, Nn, 256, 256, 0, 0);
  fill_kernel<<<cdiv(Nn * 4, 256), 256, 0, stream>>>(wsMX, NEGINF, (long)Nn * 4);
  fill_kernel<<<cdiv(Nn * 4, 256), 256, 0, stream>>>(wsDEN, 0.0f, (long)Nn * 4);
  fill_kernel<<<cdiv((long)Nn * 256, 256), 256, 0, stream>>>(wsOUT, 0.0f, (long)Nn * 256);
  gat_logit<<<cdiv(EP, 8), 256, 0, stream>>>(wsXL, wsXR, edge_attr, wsLAT, srcv, dstv, g2_we, g2_att, wsLG, wsMX, E, Nn);
  gat_expsum<<<cdiv(EP, 256), 256, 0, stream>>>(dstv, wsLG, wsMX, wsDEN, E, Nn);
  gat_scatter<<<cdiv(EP, 8), 256, 0, stream>>>(wsXL, wsLG, wsDEN, srcv, dstv, wsOUT, E, Nn);
  gat2_final<<<cdiv(Nn * 64, 256), 256, 0, stream>>>(wsOUT, g2_b, bn2_g, bn2_b, wsH0, Nn);  // h2 -> wsH0

  // ---- SAGE: agg = mean_in(h2); h3 = agg@wl + bl + h2@wr ; bn3 + relu ----
  fill_kernel<<<cdiv(Nn * 64, 256), 256, 0, stream>>>(wsH, 0.0f, (long)Nn * 64);
  sage_scatter<<<cdiv(E, 8), 256, 0, stream>>>(wsH0, srcv, dstv, wsH, E);
  sage_div<<<cdiv(Nn * 64, 256), 256, 0, stream>>>(wsH, wsCNT, Nn);
  wmma_gemm<<<dim3(4, cdiv(Nn, 64)), 128, 0, stream>>>(wsH, pSWL, sage_bl, wsH3, Nn, 64, 64, 0, 0);
  wmma_gemm<<<dim3(4, cdiv(Nn, 64)), 128, 0, stream>>>(wsH0, pSWR, nullptr, wsH3, Nn, 64, 64, 0, 1);
  bn3_relu<<<cdiv(Nn * 64, 256), 256, 0, stream>>>(wsH3, bn3_g, bn3_b, Nn);

  // ---- pooling + heads ----
  pool_kernel<<<cdiv(Nn, 8), 256, 0, stream>>>(wsH3, batch, wsMEAN, wsMAX, wsCNTG, Nn);
  graph_head<<<1, 128, 0, stream>>>(wsMEAN, wsMAX, wsCNTG, gh_w1, gh_b1, gh_w2, gh_b2, gh_w3, gh_b3, out);
  edge_head<<<cdiv(E, 256), 256, 0, stream>>>(wsH3, edge_attr, srcv, dstv, eh_w1, eh_b1, eh_w2, eh_b2, out + N_GRAPH, E);
}
